// SelfAttentionHead_42520176231011
// MI455X (gfx1250) — matrix operand order repaired and checked by reading, not compiled, not run on hardware
//
#include <hip/hip_runtime.h>
#include <hip/hip_bf16.h>

typedef __attribute__((ext_vector_type(16))) _Float16 v16h;
typedef __attribute__((ext_vector_type(8)))  _Float16 v8h;
typedef __attribute__((ext_vector_type(8)))  float    v8f;

#define T_SEQ 4096
#define CDIM  1024
#define DDIM  128
#define BATCH 4
#define VSTR  72  // LDS V^T tile row stride in halves (64 keys + 8 pad)

static __device__ __forceinline__ v16h make16(v8h lo, v8h hi) {
  v16h r;
#pragma unroll
  for (int i = 0; i < 8; i++) { r[i] = lo[i]; r[i + 8] = hi[i]; }
  return r;
}

static __device__ __forceinline__ v8f zero8() {
  v8f z;
#pragma unroll
  for (int i = 0; i < 8; i++) z[i] = 0.0f;
  return z;
}

static __device__ __forceinline__ v8f wmma_f16(v16h a, v16h b, v8f c) {
  return __builtin_amdgcn_wmma_f32_16x16x32_f16(false, a, false, b, (short)0, c,
                                                false, false);
}

// Async DMA of one 128-byte row segment (8 x b128) global -> LDS.
// INST_OFFSET adds to BOTH the LDS and the global address, and our strides
// match (16B chunks along the row), so one address pair serves all 8 ops.
static __device__ __forceinline__ void stage_row_async(const _Float16* gsrc,
                                                       unsigned ldsoff) {
  asm volatile(
      "global_load_async_to_lds_b128 %0, %1, off\n\t"
      "global_load_async_to_lds_b128 %0, %1, off offset:16\n\t"
      "global_load_async_to_lds_b128 %0, %1, off offset:32\n\t"
      "global_load_async_to_lds_b128 %0, %1, off offset:48\n\t"
      "global_load_async_to_lds_b128 %0, %1, off offset:64\n\t"
      "global_load_async_to_lds_b128 %0, %1, off offset:80\n\t"
      "global_load_async_to_lds_b128 %0, %1, off offset:96\n\t"
      "global_load_async_to_lds_b128 %0, %1, off offset:112"
      :
      : "v"(ldsoff), "v"(gsrc)
      : "memory");
}

// ---------------------------------------------------------------------------
// Kernel 1: convert Wq/Wk/Wv (fp32 [C,D]) -> WT f16 [3][D][C] (transposed)
// ---------------------------------------------------------------------------
__global__ void wt_convert_kernel(const float* __restrict__ Wq,
                                  const float* __restrict__ Wk,
                                  const float* __restrict__ Wv,
                                  _Float16* __restrict__ WT) {
  int idx = blockIdx.x * blockDim.x + threadIdx.x;
  const int total = 3 * CDIM * DDIM;
  if (idx >= total) return;
  int mat = idx / (CDIM * DDIM);
  int rem = idx % (CDIM * DDIM);
  int d = rem / CDIM;
  int c = rem % CDIM;
  const float* W = (mat == 0) ? Wq : (mat == 1) ? Wk : Wv;
  WT[idx] = (_Float16)W[c * DDIM + d];
}

// ---------------------------------------------------------------------------
// Kernel 2: QKV projection GEMM.  grid = (M/64, 3), block = 128 (4 waves).
//   mat 0 (Q): row-major [B*T][128], pre-scaled by 1/sqrt(D)
//   mat 1 (K): row-major [B*T][128]
//   mat 2 (V): transposed per batch: VT[b][d=128][t=4096]
// ---------------------------------------------------------------------------
__global__ __launch_bounds__(128) void qkv_gemm_kernel(
    const float* __restrict__ x, const _Float16* __restrict__ WT,
    _Float16* __restrict__ qkv) {
  const int lane = threadIdx.x & 31;
  const int wav  = threadIdx.x >> 5;
  const int hf   = lane >> 4;
  const int cl   = lane & 15;
  const int mat  = blockIdx.y;
  const int r0   = blockIdx.x * 64 + wav * 16;

  const _Float16* Wb   = WT + (size_t)mat * (DDIM * CDIM);
  const float*    xrow = x + (size_t)(r0 + cl) * CDIM;

  v8f acc[8];
#pragma unroll
  for (int nt = 0; nt < 8; nt++) acc[nt] = zero8();

  for (int c0 = 0; c0 < CDIM; c0 += 32) {
    v8h lo, hi;
    const float* p0 = xrow + c0 + hf * 8;
    const float* p1 = xrow + c0 + 16 + hf * 8;
#pragma unroll
    for (int i = 0; i < 8; i++) {
      lo[i] = (_Float16)p0[i];
      hi[i] = (_Float16)p1[i];
    }
    v16h a = make16(lo, hi);
#pragma unroll
    for (int nt = 0; nt < 8; nt++) {
      const _Float16* bp = Wb + (size_t)(nt * 16 + cl) * CDIM + c0 + hf * 8;
      v16h b = make16(*(const v8h*)bp, *(const v8h*)(bp + 16));
      acc[nt] = wmma_f16(a, b, acc[nt]);
    }
  }

  const size_t mstride = (size_t)BATCH * T_SEQ * DDIM;
  const float oscale = (mat == 0) ? 0.08838834764831845f : 1.0f;

  if (mat < 2) {
    _Float16* dst = qkv + (size_t)mat * mstride;
#pragma unroll
    for (int nt = 0; nt < 8; nt++)
#pragma unroll
      for (int v = 0; v < 8; v++)
        dst[(size_t)(r0 + v + 8 * hf) * DDIM + nt * 16 + cl] =
            (_Float16)(acc[nt][v] * oscale);
  } else {
    const int b2 = r0 / T_SEQ;
    const int t0 = r0 - b2 * T_SEQ;
    _Float16* dst = qkv + 2 * mstride + (size_t)b2 * DDIM * T_SEQ;
#pragma unroll
    for (int nt = 0; nt < 8; nt++)
#pragma unroll
      for (int v = 0; v < 8; v++)
        dst[(size_t)(nt * 16 + cl) * T_SEQ + t0 + v + 8 * hf] =
            (_Float16)acc[nt][v];
  }
}

// ---------------------------------------------------------------------------
// Attention block processor.  NCT = # of 16-key column tiles (4 or 2).
// V B-operand comes from (vbase, vstr, vkoff): LDS tile or global V^T.
// ---------------------------------------------------------------------------
template <int NCT, bool MASK>
static __device__ __forceinline__ void attn_block(
    int kb, int q0, int hf, int cl, const _Float16* __restrict__ K,
    const _Float16* vbase, int vstr, int vkoff, const v16h* qa,
    float* __restrict__ sw, v8f* o, v8f& lacc, float& mrow,
    const v16h& ones) {
  constexpr int STR = NCT * 16 + 4;  // f32 row stride (keeps 16B alignment)

  // 1) scores: S = Q K^T  (NCT x 4 WMMAs), K B-operand from global/L2
  v8f s[NCT];
#pragma unroll
  for (int j = 0; j < NCT; j++) s[j] = zero8();
#pragma unroll
  for (int st = 0; st < 4; st++) {
#pragma unroll
    for (int j = 0; j < NCT; j++) {
      const _Float16* kp =
          K + (size_t)(kb + j * 16 + cl) * DDIM + st * 32 + hf * 8;
      v16h bk = make16(*(const v8h*)kp, *(const v8h*)(kp + 16));
      s[j] = wmma_f16(qa[st], bk, s[j]);
    }
  }

  // 2) spill S (C-layout) to wave-private LDS (in-order per wave)
#pragma unroll
  for (int j = 0; j < NCT; j++)
#pragma unroll
    for (int v = 0; v < 8; v++)
      sw[(v + 8 * hf) * STR + j * 16 + cl] = s[j][v];

  // 3) read back by rows (lane = row cl, chunks in A-operand order)
  float xv[NCT * 8];
#pragma unroll
  for (int a = 0; a < NCT / 2; a++) {
    const float* rp = sw + cl * STR + a * 32 + hf * 8;
    float4 f0 = *(const float4*)(rp);
    float4 f1 = *(const float4*)(rp + 4);
    float4 f2 = *(const float4*)(rp + 16);
    float4 f3 = *(const float4*)(rp + 20);
    xv[a * 16 + 0] = f0.x;  xv[a * 16 + 1] = f0.y;
    xv[a * 16 + 2] = f0.z;  xv[a * 16 + 3] = f0.w;
    xv[a * 16 + 4] = f1.x;  xv[a * 16 + 5] = f1.y;
    xv[a * 16 + 6] = f1.z;  xv[a * 16 + 7] = f1.w;
    xv[a * 16 + 8]  = f2.x; xv[a * 16 + 9]  = f2.y;
    xv[a * 16 + 10] = f2.z; xv[a * 16 + 11] = f2.w;
    xv[a * 16 + 12] = f3.x; xv[a * 16 + 13] = f3.y;
    xv[a * 16 + 14] = f3.z; xv[a * 16 + 15] = f3.w;
  }

  // 4) causal mask (diagonal blocks only)
  if (MASK) {
    const int qi = q0 + cl;
#pragma unroll
    for (int a = 0; a < NCT / 2; a++)
#pragma unroll
      for (int i = 0; i < 16; i++) {
        int c = kb + a * 32 + ((i < 8) ? (hf * 8 + i) : (16 + hf * 8 + i - 8));
        if (c > qi) xv[a * 16 + i] = -1.0e10f;
      }
  }

  // 5) row max: in-lane reduce + one cross-half shuffle
  float mx = xv[0];
#pragma unroll
  for (int i = 1; i < NCT * 8; i++) mx = fmaxf(mx, xv[i]);
  mx = fmaxf(mx, __shfl_xor(mx, 16, 32));
  const float mn = fmaxf(mrow, mx);
  const float alpha = __expf(mrow - mn);
  mrow = mn;

  // 6) exponentiate in-lane; result IS the P A-operand
  v16h pa[NCT / 2];
#pragma unroll
  for (int a = 0; a < NCT / 2; a++)
#pragma unroll
    for (int i = 0; i < 16; i++)
      pa[a][i] = (_Float16)__expf(xv[a * 16 + i] - mn);

  // 7) broadcast alpha back to C-layout rows (8 bpermute)
  float av[8];
#pragma unroll
  for (int v = 0; v < 8; v++) av[v] = __shfl(alpha, v + 8 * hf, 32);

  // 8) rescale O and the row-sum accumulator
#pragma unroll
  for (int nt = 0; nt < 8; nt++)
#pragma unroll
    for (int v = 0; v < 8; v++) o[nt][v] *= av[v];
#pragma unroll
  for (int v = 0; v < 8; v++) lacc[v] *= av[v];

  // 9) O += P*V; row sums via P*ones WMMA
#pragma unroll
  for (int a = 0; a < NCT / 2; a++) {
#pragma unroll
    for (int nt = 0; nt < 8; nt++) {
      const _Float16* vp =
          vbase + (size_t)(nt * 16 + cl) * vstr + vkoff + a * 32 + hf * 8;
      v16h vb = make16(*(const v8h*)vp, *(const v8h*)(vp + 16));
      o[nt] = wmma_f16(pa[a], vb, o[nt]);
    }
    lacc = wmma_f16(pa[a], ones, lacc);
  }
}

// ---------------------------------------------------------------------------
// Kernel 3: causal flash attention.  grid = (T/64, B), block = 128 (4 waves).
// Main loop (uniform trip count): V^T tile double-buffered in LDS via
// async-to-LDS DMA; K streamed from global/L2.  Epilogue: masked 32-key
// blocks per-wave with global V^T.
// ---------------------------------------------------------------------------
__global__ __launch_bounds__(128) void attn_kernel(
    const _Float16* __restrict__ qkv, float* __restrict__ out) {
  __shared__ float    sld[4 * 16 * 68];       // per-wave f32 S tiles
  __shared__ _Float16 vbuf[2][DDIM * VSTR];   // double-buffered V^T tile

  const int tid  = threadIdx.x;
  const int lane = tid & 31;
  const int wav  = tid >> 5;
  const int hf   = lane >> 4;
  const int cl   = lane & 15;
  const int b     = blockIdx.y;
  const int qtile = blockIdx.x;
  const int q0    = qtile * 64 + wav * 16;

  const size_t mstride = (size_t)BATCH * T_SEQ * DDIM;
  const _Float16* Q  = qkv + (size_t)b * T_SEQ * DDIM;
  const _Float16* K  = qkv + mstride + (size_t)b * T_SEQ * DDIM;
  const _Float16* VT = qkv + 2 * mstride + (size_t)b * DDIM * T_SEQ;

  // Q A-tiles (16 queries x 128 d), pre-scaled by 1/sqrt(D) in kernel 2.
  v16h qa[4];
  const _Float16* qrow = Q + (size_t)(q0 + cl) * DDIM;
#pragma unroll
  for (int st = 0; st < 4; st++) {
    const _Float16* p0 = qrow + st * 32 + hf * 8;
    const _Float16* p1 = qrow + st * 32 + 16 + hf * 8;
    qa[st] = make16(*(const v8h*)p0, *(const v8h*)p1);
  }

  v16h ones;
#pragma unroll
  for (int i = 0; i < 16; i++) ones[i] = (_Float16)1.0f;

  v8f o[8];
#pragma unroll
  for (int nt = 0; nt < 8; nt++) o[nt] = zero8();
  v8f lacc = zero8();
  float mrow = -3.0e38f;

  float* sw = sld + wav * 16 * 68;

  // LDS byte offsets of the two V^T buffers for this thread's DMA row (d=tid)
  const unsigned vlds0 = (unsigned)(size_t)(&vbuf[0][tid * VSTR]);
  const unsigned vlds1 = (unsigned)(size_t)(&vbuf[1][tid * VSTR]);

  // prologue: stage block 0 V^T tile (row d = tid, 64 keys = 128 bytes)
  if (qtile > 0) stage_row_async(VT + (size_t)tid * T_SEQ, vlds0);

  // main loop: qtile full 64-key blocks, uniform across all 4 waves
  for (int blk = 0; blk < qtile; ++blk) {
    const int kb = blk * 64;
    __builtin_prefetch(K + (size_t)(kb + 64 + lane) * DDIM, 0, 0);
    __builtin_prefetch(K + (size_t)(kb + 96 + lane) * DDIM, 0, 0);

    if (blk + 1 < qtile) {  // stage next tile into the other buffer
      stage_row_async(VT + (size_t)tid * T_SEQ + kb + 64,
                      (blk & 1) ? vlds0 : vlds1);
      asm volatile("s_wait_asynccnt 0x8" ::: "memory");  // current tile done
    } else {
      asm volatile("s_wait_asynccnt 0x0" ::: "memory");
    }
    __syncthreads();  // all waves' DMA for current tile complete

    attn_block<4, false>(kb, q0, hf, cl, K, vbuf[blk & 1], VSTR, 0, qa, sw, o,
                         lacc, mrow, ones);
    __syncthreads();  // readers done before this buffer is re-staged
  }

  // epilogue: masked 32-key diagonal blocks, per-wave causal bound
  for (int kb = qtile * 64; kb < q0 + 16; kb += 32)
    attn_block<2, true>(kb, q0, hf, cl, K, VT, T_SEQ, kb, qa, sw, o, lacc,
                        mrow, ones);

  // normalize by row sums (lacc holds rowsum in every column) and store fp32
  float* orow = out + ((size_t)b * T_SEQ + q0) * DDIM;
#pragma unroll
  for (int v = 0; v < 8; v++) {
    const float inv = 1.0f / lacc[v];
#pragma unroll
    for (int nt = 0; nt < 8; nt++)
      orow[(size_t)(v + 8 * hf) * DDIM + nt * 16 + cl] = o[nt][v] * inv;
  }
}

// ---------------------------------------------------------------------------
extern "C" void kernel_launch(void* const* d_in, const int* in_sizes, int n_in,
                              void* d_out, int out_size, void* d_ws,
                              size_t ws_size, hipStream_t stream) {
  const float* x  = (const float*)d_in[0];
  const float* Wq = (const float*)d_in[1];
  const float* Wk = (const float*)d_in[2];
  const float* Wv = (const float*)d_in[3];
  float* out = (float*)d_out;

  char* ws = (char*)d_ws;
  _Float16* WT  = (_Float16*)ws;                          // 3*128*1024 halves
  _Float16* qkv = (_Float16*)(ws + 3 * DDIM * CDIM * 2);  // Q,K row-major + VT

  const int totalW = 3 * CDIM * DDIM;
  wt_convert_kernel<<<(totalW + 255) / 256, 256, 0, stream>>>(Wq, Wk, Wv, WT);

  dim3 g2((BATCH * T_SEQ) / 64, 3);
  qkv_gemm_kernel<<<g2, 128, 0, stream>>>(x, WT, qkv);

  dim3 g3(T_SEQ / 64, BATCH);
  attn_kernel<<<g3, 128, 0, stream>>>(qkv, out);
}
